// AuxSMNet_15255723835830
// MI455X (gfx1250) — matrix operand order repaired and checked by reading, not compiled, not run on hardware
//
#include <hip/hip_runtime.h>
#include <hip/hip_bf16.h>
#include <stdint.h>

#define T_FRAMES 6
#define C_IN     64
#define IMG_H    480
#define IMG_W    640
#define NPIX     (IMG_H*IMG_W)      // 307200
#define MAPW     192
#define MPOS     (MAPW*MAPW)        // 36864
#define POSTILES (MPOS/16)          // 2304
#define PGROUPS  (POSTILES/4)       // 576 groups of 4 position-tiles (64 columns)
#define MEMC     128
#define G3       384
#define NOBJ     13
#define BN_EPS   1e-5f

typedef __attribute__((ext_vector_type(16))) __bf16 v16bf;
typedef __attribute__((ext_vector_type(8)))  float  v8f;

union Frag { uint4 q[2]; v16bf v; };

// ---------- helpers ----------
static __device__ __forceinline__ unsigned short f2bf(float f) {
  unsigned u = __float_as_uint(f);
  unsigned r = 0x7FFFu + ((u >> 16) & 1u);       // round-to-nearest-even
  return (unsigned short)((u + r) >> 16);
}
static __device__ __forceinline__ unsigned ordf(float f) {   // order-preserving map
  unsigned u = __float_as_uint(f);
  return (u & 0x80000000u) ? ~u : (u | 0x80000000u);
}
static __device__ __forceinline__ float unordf(unsigned o) {
  unsigned u = (o & 0x80000000u) ? (o ^ 0x80000000u) : ~o;
  return __uint_as_float(u);
}
#define PACK_INIT 0x007FFFFF00000000ULL   // ordf(-inf)<<32 | pix 0

// ---------- encode kernels ----------
__global__ void init_packed_kernel(unsigned long long* __restrict__ packed) {
  int p = blockIdx.x * 256 + threadIdx.x;
  if (p < MPOS) packed[p] = PACK_INIT;
}

__global__ void scatter_kernel(const int* __restrict__ wtm,
                               const unsigned char* __restrict__ outl,
                               const float* __restrict__ hgt,
                               unsigned long long* __restrict__ packed) {
  int pix = blockIdx.x * 256 + threadIdx.x;
  if (pix >= NPIX) return;
  if (outl[pix]) return;                               // outlier -> -inf (never wins)
  int fx = wtm[2 * pix + 0];
  int fy = wtm[2 * pix + 1];
  int flat = MAPW * fy + fx;
  float v = hgt[pix] + 1000.0f;
  unsigned long long pk = (((unsigned long long)ordf(v)) << 32) | (unsigned)pix;
  atomicMax(packed + flat, pk);                        // max val, ties -> max pix
}

__global__ void cells_kernel(const unsigned long long* __restrict__ packed,
                             const float* __restrict__ feat,   // [C][NPIX] this frame
                             float* __restrict__ hmap,
                             unsigned* __restrict__ obs,
                             unsigned* __restrict__ mflag,
                             unsigned short* __restrict__ xg)  // [MPOS][64] bf16
{
  int p = blockIdx.x * 256 + threadIdx.x;
  if (p >= MPOS) return;
  unsigned long long pk = packed[p];
  float cand = unordf((unsigned)(pk >> 32));
  unsigned pix = (unsigned)pk;
  if (pix >= NPIX) pix = NPIX - 1;
  bool m = cand > hmap[p];
  if (m) { hmap[p] = cand; obs[p] = 1u; }
  mflag[p] = m ? 1u : 0u;
  unsigned short* o = xg + (size_t)p * C_IN;
  #pragma unroll 4
  for (int c = 0; c < C_IN; ++c) o[c] = f2bf(feat[(size_t)c * NPIX + pix]);
}

__global__ void f32_to_bf16_kernel(const float* __restrict__ s,
                                   unsigned short* __restrict__ d, int n) {
  int i = blockIdx.x * 256 + threadIdx.x;
  if (i < n) d[i] = f2bf(s[i]);
}

__global__ void gates_kernel(const float* __restrict__ gi, const float* __restrict__ gh,
                             const float* __restrict__ bih, const float* __restrict__ bhh,
                             const unsigned* __restrict__ mflag, float* __restrict__ state) {
  int i = blockIdx.x * 256 + threadIdx.x;
  if (i >= MPOS * MEMC) return;
  int p = i >> 7, j = i & 127;
  if (!mflag[p]) return;                                // unobserved cells keep state
  float gr  = gi[(size_t)j * MPOS + p]        + bih[j]        + gh[(size_t)j * MPOS + p]        + bhh[j];
  float gz  = gi[(size_t)(128 + j) * MPOS + p] + bih[128 + j] + gh[(size_t)(128 + j) * MPOS + p] + bhh[128 + j];
  float gin = gi[(size_t)(256 + j) * MPOS + p] + bih[256 + j];
  float ghn = gh[(size_t)(256 + j) * MPOS + p] + bhh[256 + j];
  float r = 1.f / (1.f + __expf(-gr));
  float z = 1.f / (1.f + __expf(-gz));
  float n = tanhf(gin + r * ghn);
  float h = state[i];
  state[i] = (1.f - z) * n + z * h;
}

// ---------- weight prepack: [row][chunk=(kh*ks+kw)*cinBlocks+cb][32] bf16 ----------
__global__ void prepack_kernel(const float* __restrict__ w, unsigned short* __restrict__ pk,
                               int rowsPad, int coutReal, int cin, int cinPad, int ks) {
  int cinBlocks = cinPad >> 5;
  int nchunks = cinBlocks * ks * ks;
  long long total = (long long)rowsPad * nchunks * 32;
  long long i = (long long)blockIdx.x * 256 + threadIdx.x;
  if (i >= total) return;
  int row = (int)(i / ((long long)nchunks * 32));
  int rem = (int)(i % ((long long)nchunks * 32));
  int chunk = rem >> 5, c = rem & 31;
  int tap = chunk / cinBlocks, cb = chunk % cinBlocks;
  int kh = tap / ks, kw = tap % ks;
  int ci = cb * 32 + c;
  float v = 0.f;
  if (row < coutReal && ci < cin)
    v = w[(((size_t)row * cin + ci) * ks + kh) * ks + kw];
  pk[i] = f2bf(v);
}

// ---------- generic implicit-GEMM conv / GEMM via bf16 WMMA ----------
// 2x4 register blocking: each wave computes 32 output rows x 64 output columns.
// act: [MPOS][cinPad] bf16 (channel-last), wpk: packed weights, out: [rowsPad][MPOS] f32
__global__ __launch_bounds__(256) void conv_wmma_kernel(
    const unsigned short* __restrict__ act,
    const unsigned short* __restrict__ wpk,
    float* __restrict__ out,
    int ocGroups,          // rowsPad / 32
    int cinPad, int ks, int pad, int nchunks) {
  int wave = blockIdx.x * 8 + (threadIdx.x >> 5);
  if (wave >= ocGroups * PGROUPS) return;              // wave-uniform, EXEC stays full
  int og = wave / PGROUPS;
  int pg = wave % PGROUPS;
  int lane = threadIdx.x & 31;
  int lx = lane & 15;
  int hi = lane >> 4;                                   // 0 or 1
  int pos0 = pg * 64;
  int y = pos0 / MAPW;                                  // 64 | 192 -> one row per group
  int xbase = pos0 % MAPW;
  int xl = xbase + lx;
  int cinBlocks = cinPad >> 5;
  int rowA = og * 32 + lx;
  const uint4* wq = (const uint4*)wpk;                  // 4 x uint4 per 32-elem chunk
  size_t arow = (size_t)nchunks * 4;                    // uint4 per weight row

  const v8f vz = {0.f,0.f,0.f,0.f,0.f,0.f,0.f,0.f};
  v8f acc[2][4];
  #pragma unroll
  for (int o = 0; o < 2; ++o)
    #pragma unroll
    for (int s = 0; s < 4; ++s) acc[o][s] = vz;

  for (int kh = 0; kh < ks; ++kh) {
    int yy = y + kh - pad;
    if ((unsigned)yy >= (unsigned)MAPW) continue;       // uniform across wave
    if (ks > 1) {                                       // prefetch next activation row
      int yn = yy + 1;
      if (yn < MAPW)
        __builtin_prefetch(act + ((size_t)(yn * MAPW + xbase)) * cinPad, 0, 3);
    }
    int rowbase = yy * MAPW;
    for (int kw = 0; kw < ks; ++kw) {
      int xx0 = xl + kw - pad;
      int tap = kh * ks + kw;
      for (int cb = 0; cb < cinBlocks; ++cb) {
        int chunk = tap * cinBlocks + cb;
        size_t aq = ((size_t)rowA * nchunks + chunk) * 4 + 2 * hi;
        Frag A0, A1;
        A0.q[0] = wq[aq];              A0.q[1] = wq[aq + 1];
        A1.q[0] = wq[aq + 16 * arow];  A1.q[1] = wq[aq + 16 * arow + 1];
        Frag B[4];
        #pragma unroll
        for (int s = 0; s < 4; ++s) {
          int xx = xx0 + s * 16;
          if ((unsigned)xx < (unsigned)MAPW) {
            const uint4* bq = (const uint4*)(act + ((size_t)(rowbase + xx)) * cinPad
                                             + (size_t)cb * 32 + (hi ? 16 : 0));
            B[s].q[0] = bq[0]; B[s].q[1] = bq[1];
          } else {
            B[s].q[0] = make_uint4(0, 0, 0, 0);
            B[s].q[1] = make_uint4(0, 0, 0, 0);
          }
        }
        #pragma unroll
        for (int s = 0; s < 4; ++s) {
          acc[0][s] = __builtin_amdgcn_wmma_f32_16x16x32_bf16(
              false, A0.v, false, B[s].v, (short)0, acc[0][s], false, false);
          acc[1][s] = __builtin_amdgcn_wmma_f32_16x16x32_bf16(
              false, A1.v, false, B[s].v, (short)0, acc[1][s], false, false);
        }
      }
    }
  }
  #pragma unroll
  for (int o = 0; o < 2; ++o) {
    #pragma unroll
    for (int s = 0; s < 4; ++s) {
      float* op = out + (size_t)(og * 32 + o * 16 + hi * 8) * MPOS + pos0 + s * 16 + lx;
      #pragma unroll
      for (int r = 0; r < 8; ++r) op[(size_t)r * MPOS] = acc[o][s][r];
    }
  }
}

// ---------- BatchNorm (training-mode batch stats) ----------
__global__ __launch_bounds__(256) void bn_stats_kernel(const float* __restrict__ x,
                                                       float* __restrict__ stats) {
  int c = blockIdx.x;
  const float* xc = x + (size_t)c * MPOS;
  float s = 0.f, ss = 0.f;
  for (int i = threadIdx.x; i < MPOS; i += 256) { float v = xc[i]; s += v; ss += v * v; }
  __shared__ float sh[256], sh2[256];
  sh[threadIdx.x] = s; sh2[threadIdx.x] = ss;
  __syncthreads();
  for (int o = 128; o > 0; o >>= 1) {
    if (threadIdx.x < o) { sh[threadIdx.x] += sh[threadIdx.x + o]; sh2[threadIdx.x] += sh2[threadIdx.x + o]; }
    __syncthreads();
  }
  if (threadIdx.x == 0) {
    float mu = sh[0] / (float)MPOS;
    float var = sh2[0] / (float)MPOS - mu * mu;
    stats[2 * c] = mu;
    stats[2 * c + 1] = rsqrtf(var + BN_EPS);
  }
}

// BN + optional residual + optional relu; writes bf16 channel-last (+ optional f32 copy)
__global__ void bn_apply_kernel(const float* __restrict__ x, const float* __restrict__ stats,
                                const float* __restrict__ g, const float* __restrict__ b,
                                const float* __restrict__ res, float* __restrict__ f32out,
                                unsigned short* __restrict__ act, int creal, int cpad, int relu) {
  long long i = (long long)blockIdx.x * 256 + threadIdx.x;
  if (i >= (long long)creal * MPOS) return;
  int c = (int)(i / MPOS), p = (int)(i % MPOS);
  float v = (x[i] - stats[2 * c]) * stats[2 * c + 1] * g[c] + b[c];
  if (res) v += res[i];
  if (relu) v = fmaxf(v, 0.f);
  if (f32out) f32out[i] = v;
  act[(size_t)p * cpad + c] = f2bf(v);
}

__global__ void finalize_kernel(const float* __restrict__ o2, const float* __restrict__ bias,
                                const unsigned* __restrict__ obs, const float* __restrict__ hmap,
                                float* __restrict__ out) {
  int p = blockIdx.x * 256 + threadIdx.x;
  if (p >= MPOS) return;
  #pragma unroll
  for (int k = 0; k < NOBJ; ++k)
    out[(size_t)k * MPOS + p] = o2[(size_t)k * MPOS + p] + bias[k];
  out[(size_t)NOBJ * MPOS + p] = obs[p] ? 1.f : 0.f;
  out[(size_t)(NOBJ + 1) * MPOS + p] = hmap[p];
}

// ---------- host ----------
extern "C" void kernel_launch(void* const* d_in, const int* in_sizes, int n_in,
                              void* d_out, int out_size, void* d_ws, size_t ws_size,
                              hipStream_t stream) {
  (void)in_sizes; (void)n_in; (void)out_size; (void)ws_size;
  const float*         features = (const float*)d_in[0];
  const int*           proj     = (const int*)d_in[1];
  const unsigned char* outliers = (const unsigned char*)d_in[2];
  const float*         heights  = (const float*)d_in[3];
  const float* w_ih = (const float*)d_in[6];
  const float* w_hh = (const float*)d_in[7];
  const float* b_ih = (const float*)d_in[8];
  const float* b_hh = (const float*)d_in[9];
  const float* o2b  = (const float*)d_in[10];
  const float* wc1=(const float*)d_in[11], *gc1=(const float*)d_in[12], *bc1=(const float*)d_in[13];
  const float* wc2=(const float*)d_in[14], *gc2=(const float*)d_in[15], *bc2=(const float*)d_in[16];
  const float* wc3=(const float*)d_in[17], *gc3=(const float*)d_in[18], *bc3=(const float*)d_in[19];
  const float* wr1=(const float*)d_in[20], *gr1=(const float*)d_in[21], *br1=(const float*)d_in[22];
  const float* wr2=(const float*)d_in[23], *gr2=(const float*)d_in[24], *br2=(const float*)d_in[25];
  const float* wr3=(const float*)d_in[26], *gr3=(const float*)d_in[27], *br3=(const float*)d_in[28];
  const float* wo1=(const float*)d_in[29], *go1=(const float*)d_in[30], *bo1=(const float*)d_in[31];
  const float* wo2=(const float*)d_in[32];

  uint8_t* base = (uint8_t*)d_ws;
  size_t off = 0;
  auto alloc = [&](size_t bytes) -> void* {
    void* p = base + off; off = (off + bytes + 255) & ~(size_t)255; return p;
  };
  // persistent / encode region
  float*              state  = (float*)alloc((size_t)MPOS * MEMC * 4);
  float*              hmap   = (float*)alloc((size_t)MPOS * 4);
  unsigned*           obs    = (unsigned*)alloc((size_t)MPOS * 4);
  unsigned long long* packed = (unsigned long long*)alloc((size_t)MPOS * 8);
  unsigned*           mflag  = (unsigned*)alloc((size_t)MPOS * 4);
  unsigned short*     wihpk  = (unsigned short*)alloc((size_t)G3 * 2 * 32 * 2);
  unsigned short*     whhpk  = (unsigned short*)alloc((size_t)G3 * 4 * 32 * 2);
  float*              stats  = (float*)alloc(2 * 128 * 4);
  unsigned short*     xg     = (unsigned short*)alloc((size_t)MPOS * 64 * 2);
  unsigned short*     hbf    = (unsigned short*)alloc((size_t)MPOS * 128 * 2); // also decoder act0
  // gate buffers (encode-only) -- decoder buffers overlay this region
  float* gi = (float*)alloc((size_t)G3 * MPOS * 4);
  float* gh = (float*)alloc((size_t)G3 * MPOS * 4);
  // decoder overlay inside [gi .. gi + 2*G3*MPOS*4)
  size_t uoff = 0;
  uint8_t* ubase = (uint8_t*)gi;
  auto ualloc = [&](size_t bytes) -> void* {
    void* p = ubase + uoff; uoff = (uoff + bytes + 255) & ~(size_t)255; return p;
  };
  float*          convf32 = (float*)ualloc((size_t)128 * MPOS * 4);
  float*          x3f32   = (float*)ualloc((size_t)48 * MPOS * 4);
  unsigned short* act1    = (unsigned short*)ualloc((size_t)MPOS * 128 * 2);
  unsigned short* act2    = (unsigned short*)ualloc((size_t)MPOS * 64 * 2);
  unsigned short* x3bf    = (unsigned short*)ualloc((size_t)MPOS * 64 * 2);
  unsigned short* r1act   = (unsigned short*)ualloc((size_t)MPOS * 32 * 2);
  unsigned short* r2act   = (unsigned short*)ualloc((size_t)MPOS * 32 * 2);
  unsigned short* resbf   = (unsigned short*)ualloc((size_t)MPOS * 64 * 2);
  unsigned short* o1act   = (unsigned short*)ualloc((size_t)MPOS * 64 * 2);
  unsigned short* c1pk = (unsigned short*)ualloc((size_t)128 * 196 * 32 * 2);
  unsigned short* c2pk = (unsigned short*)ualloc((size_t)64 * 36 * 32 * 2);
  unsigned short* c3pk = (unsigned short*)ualloc((size_t)64 * 18 * 32 * 2);
  unsigned short* r1pk = (unsigned short*)ualloc((size_t)32 * 2 * 32 * 2);
  unsigned short* r2pk = (unsigned short*)ualloc((size_t)32 * 9 * 32 * 2);
  unsigned short* r3pk = (unsigned short*)ualloc((size_t)64 * 1 * 32 * 2);
  unsigned short* o1pk = (unsigned short*)ualloc((size_t)64 * 18 * 32 * 2);
  unsigned short* o2pk = (unsigned short*)ualloc((size_t)32 * 2 * 32 * 2);

  auto conv = [&](const unsigned short* act, const unsigned short* pk, float* out,
                  int rowsPad, int cinPad, int ks, int pad) {
    int nchunks = (cinPad / 32) * ks * ks;
    int waves = (rowsPad / 32) * PGROUPS;
    conv_wmma_kernel<<<(waves + 7) / 8, 256, 0, stream>>>(act, pk, out, rowsPad / 32,
                                                          cinPad, ks, pad, nchunks);
  };
  auto prepack = [&](const float* w, unsigned short* pk, int rowsPad, int coutReal,
                     int cin, int cinPad, int ks) {
    long long tot = (long long)rowsPad * (cinPad / 32) * ks * ks * 32;
    prepack_kernel<<<(unsigned)((tot + 255) / 256), 256, 0, stream>>>(w, pk, rowsPad, coutReal, cin, cinPad, ks);
  };
  auto bn = [&](const float* x, const float* g, const float* bb, const float* res,
                float* f32out, unsigned short* act, int creal, int cpad, int relu) {
    bn_stats_kernel<<<creal, 256, 0, stream>>>(x, stats);
    long long tot = (long long)creal * MPOS;
    bn_apply_kernel<<<(unsigned)((tot + 255) / 256), 256, 0, stream>>>(x, stats, g, bb, res, f32out, act, creal, cpad, relu);
  };

  // ===== encode =====
  hipMemsetAsync(state, 0, (size_t)MPOS * MEMC * 4, stream);
  hipMemsetAsync(hmap, 0, (size_t)MPOS * 4, stream);
  hipMemsetAsync(obs, 0, (size_t)MPOS * 4, stream);
  prepack(w_ih, wihpk, G3, G3, 64, 64, 1);
  prepack(w_hh, whhpk, G3, G3, 128, 128, 1);

  for (int t = 0; t < T_FRAMES; ++t) {
    init_packed_kernel<<<MPOS / 256, 256, 0, stream>>>(packed);
    scatter_kernel<<<NPIX / 256, 256, 0, stream>>>(proj + (size_t)t * NPIX * 2,
                                                   outliers + (size_t)t * NPIX,
                                                   heights + (size_t)t * NPIX, packed);
    cells_kernel<<<MPOS / 256, 256, 0, stream>>>(packed, features + (size_t)t * C_IN * NPIX,
                                                 hmap, obs, mflag, xg);
    f32_to_bf16_kernel<<<(MPOS * MEMC) / 256, 256, 0, stream>>>(state, hbf, MPOS * MEMC);
    conv(xg, wihpk, gi, G3, 64, 1, 0);     // gi = x @ W_ih^T   (WMMA)
    conv(hbf, whhpk, gh, G3, 128, 1, 0);   // gh = h @ W_hh^T   (WMMA)
    gates_kernel<<<(MPOS * MEMC) / 256, 256, 0, stream>>>(gi, gh, b_ih, b_hh, mflag, state);
  }

  // ===== decode =====
  hipMemsetAsync(x3bf, 0, (size_t)MPOS * 64 * 2, stream);   // zero padded channels 48..63
  hipMemsetAsync(resbf, 0, (size_t)MPOS * 64 * 2, stream);
  hipMemsetAsync(o1act, 0, (size_t)MPOS * 64 * 2, stream);
  f32_to_bf16_kernel<<<(MPOS * MEMC) / 256, 256, 0, stream>>>(state, hbf, MPOS * MEMC); // act0

  prepack(wc1, c1pk, 128, 128, 128, 128, 7);
  prepack(wc2, c2pk, 64, 64, 128, 128, 3);
  prepack(wc3, c3pk, 64, 48, 64, 64, 3);
  prepack(wr1, r1pk, 32, 32, 48, 64, 1);
  prepack(wr2, r2pk, 32, 32, 32, 32, 3);
  prepack(wr3, r3pk, 64, 48, 32, 32, 1);
  prepack(wo1, o1pk, 64, 48, 48, 64, 3);
  prepack(wo2, o2pk, 32, NOBJ, 48, 64, 1);

  conv(hbf, c1pk, convf32, 128, 128, 7, 3);                // c1: 128 -> 128, 7x7
  bn(convf32, gc1, bc1, nullptr, nullptr, act1, 128, 128, 1);
  conv(act1, c2pk, convf32, 64, 128, 3, 1);                // c2: 128 -> 64
  bn(convf32, gc2, bc2, nullptr, nullptr, act2, 64, 64, 1);
  conv(act2, c3pk, convf32, 64, 64, 3, 1);                 // c3: 64 -> 48 (rows padded)
  bn(convf32, gc3, bc3, nullptr, x3f32, x3bf, 48, 64, 1);  // keep f32 for residual
  conv(x3bf, r1pk, convf32, 32, 64, 1, 0);                 // r1: 48 -> 32
  bn(convf32, gr1, br1, nullptr, nullptr, r1act, 32, 32, 1);
  conv(r1act, r2pk, convf32, 32, 32, 3, 1);                // r2: 32 -> 32
  bn(convf32, gr2, br2, nullptr, nullptr, r2act, 32, 32, 1);
  conv(r2act, r3pk, convf32, 64, 32, 1, 0);                // r3: 32 -> 48 (no act)
  bn(convf32, gr3, br3, x3f32, nullptr, resbf, 48, 64, 1); // BN + residual + relu
  conv(resbf, o1pk, convf32, 64, 64, 3, 1);                // o1: 48 -> 48
  bn(convf32, go1, bo1, nullptr, nullptr, o1act, 48, 64, 1);
  conv(o1act, o2pk, convf32, 32, 64, 1, 0);                // o2: 48 -> 13 (padded 32)
  finalize_kernel<<<MPOS / 256, 256, 0, stream>>>(convf32, o2b, obs, hmap, (float*)d_out);
}
